// DilatedAttention_17927193493837
// MI455X (gfx1250) — matrix operand order repaired and checked by reading, not compiled, not run on hardware
//
#include <hip/hip_runtime.h>
#include <hip/hip_bf16.h>

typedef __attribute__((ext_vector_type(16))) __bf16 v16bf;
typedef __attribute__((ext_vector_type(8)))  __bf16 v8bf;
typedef __attribute__((ext_vector_type(8)))  float  v8f;

#define S_LEN   2048
#define D_MODEL 1024
#define NH      16
#define HD      64

static __device__ __forceinline__ v8bf cvt8(v8f x) {
  return __builtin_convertvector(x, v8bf);        // native v_cvt_pk_bf16_f32 path
}
static __device__ __forceinline__ v8f zero_v8f() {
  v8f z;
#pragma unroll
  for (int i = 0; i < 8; ++i) z[i] = 0.0f;
  return z;
}
// dilated mask: same 64-segment, or 0<diff<=256 with diff%4==0
static __device__ __forceinline__ bool dil_allowed(int i, int j) {
  if ((i >> 6) == (j >> 6)) return true;
  int d = i - j; d = d < 0 ? -d : d;
  return (d > 0) && ((d & 3) == 0) && (d <= 256);
}

// ---------------- weight transpose + bf16 convert:  wt[n][k] = w[k][n] ----------------
__global__ __launch_bounds__(256) void wtrans_kernel(const float* __restrict__ w,
                                                     __bf16* __restrict__ wt) {
  __shared__ float tile[32][33];
  const int k0 = blockIdx.x * 32, n0 = blockIdx.y * 32;
#pragma unroll
  for (int i = threadIdx.y; i < 32; i += 8)
    tile[i][threadIdx.x] = w[(size_t)(k0 + i) * D_MODEL + n0 + threadIdx.x];
  __syncthreads();
#pragma unroll
  for (int i = threadIdx.y; i < 32; i += 8)
    wt[(size_t)(n0 + i) * D_MODEL + k0 + threadIdx.x] = (__bf16)tile[threadIdx.x][i];
}

// ---------------- QKV projection: per-wave 32x64 tile, bf16 WMMA, f32 accum ----------------
// z=0 -> q row-major [b][s][h*64+d], z=1 -> k row-major, z=2 -> v transposed [b][h][d][s]
__global__ __launch_bounds__(256) void qkv_proj_kernel(
    const float* __restrict__ hs,
    const __bf16* __restrict__ wtq, const __bf16* __restrict__ wtk, const __bf16* __restrict__ wtv,
    __bf16* __restrict__ qb, __bf16* __restrict__ kbuf, __bf16* __restrict__ vt) {
  const int z    = blockIdx.z;
  const __bf16* wt = (z == 0) ? wtq : ((z == 1) ? wtk : wtv);
  const int wave = threadIdx.x >> 5;
  const int lane = threadIdx.x & 31;
  const int half = lane >> 4;
  const int lan  = lane & 15;
  const int kb8  = half * 8;
  const int m0 = (blockIdx.x * 8 + wave) * 32;   // 128 M tiles of 32 rows
  const int n0 = blockIdx.y * 64;                // 16 N tiles

  v8f oc[2][4];
#pragma unroll
  for (int mh = 0; mh < 2; ++mh)
#pragma unroll
    for (int j = 0; j < 4; ++j) oc[mh][j] = zero_v8f();

  const float* arow0 = hs + (size_t)(m0 + lan) * D_MODEL;
  const float* arow1 = hs + (size_t)(m0 + 16 + lan) * D_MODEL;
  for (int kk = 0; kk < D_MODEL; kk += 32) {
    __builtin_prefetch(arow0 + kk + 32, 0, 0);   // stream A one k-tile ahead
    __builtin_prefetch(arow1 + kk + 32, 0, 0);
    union { v16bf v; v8bf h2[2]; } ua[2];        // A frag: K = 16*half..+15
#pragma unroll
    for (int mh = 0; mh < 2; ++mh) {
      const float* p = (mh ? arow1 : arow0) + kk + 16 * half;
      ua[mh].h2[0] = cvt8(*(const v8f*)(p));
      ua[mh].h2[1] = cvt8(*(const v8f*)(p + 8));
    }
#pragma unroll
    for (int j = 0; j < 4; ++j) {
      const v16bf bmat = *(const v16bf*)(wt + (size_t)(n0 + 16 * j + lan) * D_MODEL +
                                         kk + 16 * half);
#pragma unroll
      for (int mh = 0; mh < 2; ++mh)            // B-fragment reused for both M halves
        oc[mh][j] = __builtin_amdgcn_wmma_f32_16x16x32_bf16(false, ua[mh].v, false, bmat,
                                                            (short)0, oc[mh][j], false, false);
    }
  }

  if (z < 2) {
    __bf16* outp = (z == 0) ? qb : kbuf;
#pragma unroll
    for (int mh = 0; mh < 2; ++mh)
#pragma unroll
      for (int j = 0; j < 4; ++j) {
        const int c = n0 + 16 * j + lan;
#pragma unroll
        for (int r = 0; r < 8; ++r)
          outp[(size_t)(m0 + 16 * mh + r + 8 * half) * D_MODEL + c] = (__bf16)oc[mh][j][r];
      }
  } else {
    const int b = m0 >> 11;                    // S_LEN = 2048 rows per batch
#pragma unroll
    for (int mh = 0; mh < 2; ++mh) {
      const int s = (m0 & (S_LEN - 1)) + 16 * mh + 8 * half;
#pragma unroll
      for (int j = 0; j < 4; ++j) {
        const int c = n0 + 16 * j + lan;
        const int h = c >> 6, d = c & 63;
        *(v8bf*)(vt + (size_t)((b * NH + h) * HD + d) * S_LEN + s) = cvt8(oc[mh][j]);
      }
    }
  }
}

// ---------------- flash dilated attention: one wave per (b,h,q-tile-of-16) ----------------
__global__ __launch_bounds__(256) void dil_attn_kernel(
    const __bf16* __restrict__ qb, const __bf16* __restrict__ kbuf,
    const __bf16* __restrict__ vt, __bf16* __restrict__ ob) {
  __shared__ __bf16 pbuf[8][16][32];   // per-wave P tile staging (D-layout -> A-layout)
  const int wave = threadIdx.x >> 5;
  const int lane = threadIdx.x & 31;
  const int half = lane >> 4;
  const int lan  = lane & 15;
  const int kb8  = half * 8;
  const int gw = blockIdx.x * 8 + wave;        // 4096 waves total
  const int qt = gw & 127;
  const int h  = (gw >> 7) & (NH - 1);
  const int b  = gw >> 11;
  const int i0 = qt * 16;

  const __bf16* qbase = qb   + (size_t)b * S_LEN * D_MODEL + (size_t)h * HD;
  const __bf16* kbase = kbuf + (size_t)b * S_LEN * D_MODEL + (size_t)h * HD;
  const __bf16* vbase = vt   + (size_t)((b * NH + h) * HD) * S_LEN;

  // Q A-fragments (16x64 = two 16x32 frags)
  v16bf qa[2];
  {
    const __bf16* qrow = qbase + (size_t)(i0 + lan) * D_MODEL;
#pragma unroll
    for (int f = 0; f < 2; ++f) {
      union { v16bf v; v8bf h2[2]; } u;
      u.h2[0] = *(const v8bf*)(qrow + f * 32 + 16 * half);
      u.h2[1] = *(const v8bf*)(qrow + f * 32 + 16 * half + 8);
      qa[f] = u.v;
    }
  }

  float mrow[8], lrow[8];
  v8f oc[4];
#pragma unroll
  for (int r = 0; r < 8; ++r) { mrow[r] = -1.0e30f; lrow[r] = 0.0f; }
#pragma unroll
  for (int j = 0; j < 4; ++j) oc[j] = zero_v8f();

  int lo = (i0 - 256) & ~31; if (lo < 0) lo = 0;
  int hi = (i0 + 272 + 31) & ~31; if (hi > S_LEN) hi = S_LEN;

  const float SCL_L2E = 0.125f * 1.4426950408889634f;  // Dh^-0.5 * log2(e), exp2 domain

  for (int ks = lo; ks < hi; ks += 32) {
    // ---- S = Q K^T for 32 keys (two 16-wide N halves, K=64 via 2 chained WMMAs) ----
    v8f sc[2];
#pragma unroll
    for (int n2 = 0; n2 < 2; ++n2) {
      v8f acc = zero_v8f();
#pragma unroll
      for (int f = 0; f < 2; ++f) {
        const v16bf* kp = (const v16bf*)(kbase + (size_t)(ks + n2 * 16 + lan) * D_MODEL +
                                         f * 32 + 16 * half);
        acc = __builtin_amdgcn_wmma_f32_16x16x32_bf16(false, qa[f], false, *kp, (short)0,
                                                      acc, false, false);
      }
      sc[n2] = acc;
    }
    // ---- scale + dilated mask (exp2 domain, finite sentinel avoids inf-inf) ----
#pragma unroll
    for (int n2 = 0; n2 < 2; ++n2) {
      const int j = ks + n2 * 16 + lan;
#pragma unroll
      for (int r = 0; r < 8; ++r) {
        const int i = i0 + r + 8 * half;
        const float t = sc[n2][r] * SCL_L2E;
        sc[n2][r] = dil_allowed(i, j) ? t : -1.0e30f;
      }
    }
    // ---- online softmax: row max over 16 lanes of this half ----
    float bm[8];
#pragma unroll
    for (int r = 0; r < 8; ++r) bm[r] = fmaxf(sc[0][r], sc[1][r]);
#pragma unroll
    for (int m = 1; m < 16; m <<= 1) {
#pragma unroll
      for (int r = 0; r < 8; ++r) bm[r] = fmaxf(bm[r], __shfl_xor(bm[r], m, 32));
    }
    float alpha[8], rs[8];
#pragma unroll
    for (int r = 0; r < 8; ++r) {
      const float mn = fmaxf(mrow[r], bm[r]);
      alpha[r] = exp2f(mrow[r] - mn);
      mrow[r] = mn;
    }
#pragma unroll
    for (int r = 0; r < 8; ++r) {
      const float p0 = exp2f(sc[0][r] - mrow[r]);
      const float p1 = exp2f(sc[1][r] - mrow[r]);
      sc[0][r] = p0; sc[1][r] = p1;
      rs[r] = p0 + p1;
    }
#pragma unroll
    for (int m = 1; m < 16; m <<= 1) {
#pragma unroll
      for (int r = 0; r < 8; ++r) rs[r] += __shfl_xor(rs[r], m, 32);
    }
#pragma unroll
    for (int r = 0; r < 8; ++r) lrow[r] = lrow[r] * alpha[r] + rs[r];
#pragma unroll
    for (int j = 0; j < 4; ++j)
#pragma unroll
      for (int r = 0; r < 8; ++r) oc[j][r] *= alpha[r];

    // ---- P: D-layout -> A-layout via LDS (intra-wave LDS is in-order, no barrier) ----
#pragma unroll
    for (int r = 0; r < 8; ++r) {
      pbuf[wave][r + 8 * half][lan]      = (__bf16)sc[0][r];
      pbuf[wave][r + 8 * half][16 + lan] = (__bf16)sc[1][r];
    }
    asm volatile("" ::: "memory");
    union { v16bf v; v8bf h2[2]; } pa;
    pa.h2[0] = *(const v8bf*)&pbuf[wave][lan][16 * half];
    pa.h2[1] = *(const v8bf*)&pbuf[wave][lan][16 * half + 8];

    // ---- O += P * V  (V^T layout gives contiguous 32B B-frag loads) ----
#pragma unroll
    for (int j = 0; j < 4; ++j) {
      const v16bf* vp = (const v16bf*)(vbase + (size_t)(j * 16 + lan) * S_LEN + ks + 16 * half);
      oc[j] = __builtin_amdgcn_wmma_f32_16x16x32_bf16(false, pa.v, false, *vp, (short)0,
                                                      oc[j], false, false);
    }
  }

  float inv[8];
#pragma unroll
  for (int r = 0; r < 8; ++r) inv[r] = 1.0f / lrow[r];   // diagonal always unmasked -> l > 0
  __bf16* obase = ob + (size_t)b * S_LEN * D_MODEL + (size_t)h * HD;
#pragma unroll
  for (int j = 0; j < 4; ++j) {
#pragma unroll
    for (int r = 0; r < 8; ++r) {
      const int i = i0 + r + 8 * half;
      obase[(size_t)i * D_MODEL + j * 16 + lan] = (__bf16)(oc[j][r] * inv[r]);
    }
  }
}

// ---------------- output projection: attn(bf16) @ w_o -> f32, per-wave 32x64 ----------------
__global__ __launch_bounds__(256) void out_proj_kernel(
    const __bf16* __restrict__ ab, const __bf16* __restrict__ wto, float* __restrict__ out) {
  const int wave = threadIdx.x >> 5;
  const int lane = threadIdx.x & 31;
  const int half = lane >> 4;
  const int lan  = lane & 15;
  const int kb8  = half * 8;
  const int m0 = (blockIdx.x * 8 + wave) * 32;
  const int n0 = blockIdx.y * 64;

  v8f oc[2][4];
#pragma unroll
  for (int mh = 0; mh < 2; ++mh)
#pragma unroll
    for (int j = 0; j < 4; ++j) oc[mh][j] = zero_v8f();

  const __bf16* arow0 = ab + (size_t)(m0 + lan) * D_MODEL;
  const __bf16* arow1 = ab + (size_t)(m0 + 16 + lan) * D_MODEL;
  for (int kk = 0; kk < D_MODEL; kk += 32) {
    __builtin_prefetch(arow0 + kk + 32, 0, 0);
    __builtin_prefetch(arow1 + kk + 32, 0, 0);
    union { v16bf v; v8bf h2[2]; } ua[2];
#pragma unroll
    for (int mh = 0; mh < 2; ++mh) {
      const __bf16* p = (mh ? arow1 : arow0) + kk + 16 * half;
      ua[mh].h2[0] = *(const v8bf*)(p);
      ua[mh].h2[1] = *(const v8bf*)(p + 8);
    }
#pragma unroll
    for (int j = 0; j < 4; ++j) {
      const v16bf bmat = *(const v16bf*)(wto + (size_t)(n0 + 16 * j + lan) * D_MODEL +
                                         kk + 16 * half);
#pragma unroll
      for (int mh = 0; mh < 2; ++mh)
        oc[mh][j] = __builtin_amdgcn_wmma_f32_16x16x32_bf16(false, ua[mh].v, false, bmat,
                                                            (short)0, oc[mh][j], false, false);
    }
  }
#pragma unroll
  for (int mh = 0; mh < 2; ++mh)
#pragma unroll
    for (int j = 0; j < 4; ++j) {
      const int c = n0 + 16 * j + lan;
#pragma unroll
      for (int r = 0; r < 8; ++r)
        out[(size_t)(m0 + 16 * mh + r + 8 * half) * D_MODEL + c] = oc[mh][j][r];
    }
}

extern "C" void kernel_launch(void* const* d_in, const int* in_sizes, int n_in,
                              void* d_out, int out_size, void* d_ws, size_t ws_size,
                              hipStream_t stream) {
  (void)in_sizes; (void)n_in; (void)out_size; (void)ws_size;
  const float* hs = (const float*)d_in[0];
  const float* wq = (const float*)d_in[1];
  const float* wk = (const float*)d_in[2];
  const float* wv = (const float*)d_in[3];
  const float* wo = (const float*)d_in[4];
  float* out = (float*)d_out;

  const size_t WE = (size_t)D_MODEL * D_MODEL;       // 1M elems per weight
  const size_t AE = (size_t)2 * S_LEN * D_MODEL;     // 4M elems per activation
  __bf16* wtq = (__bf16*)d_ws;
  __bf16* wtk = wtq + WE;
  __bf16* wtv = wtk + WE;
  __bf16* wto = wtv + WE;
  __bf16* qb  = wto + WE;
  __bf16* kb  = qb + AE;
  __bf16* vt  = kb + AE;   // [b][h][d][s]
  __bf16* ab  = vt + AE;   // total 20M bf16 = 40 MB of d_ws

  dim3 tb(32, 8);
  dim3 tg(D_MODEL / 32, D_MODEL / 32);
  wtrans_kernel<<<tg, tb, 0, stream>>>(wq, wtq);
  wtrans_kernel<<<tg, tb, 0, stream>>>(wk, wtk);
  wtrans_kernel<<<tg, tb, 0, stream>>>(wv, wtv);
  wtrans_kernel<<<tg, tb, 0, stream>>>(wo, wto);

  qkv_proj_kernel<<<dim3(16, 16, 3), 256, 0, stream>>>(hs, wtq, wtk, wtv, qb, kb, vt);
  dil_attn_kernel<<<512, 256, 0, stream>>>(qb, kb, vt, ab);
  out_proj_kernel<<<dim3(16, 16), 256, 0, stream>>>(ab, wto, out);
}
